// PyTorchGemma2AttentionWrapper_42726334660777
// MI455X (gfx1250) — matrix operand order repaired and checked by reading, not compiled, not run on hardware
//
#include <hip/hip_runtime.h>
#include <hip/hip_bf16.h>
#include <math.h>

typedef __bf16 bf16_t;
typedef __attribute__((ext_vector_type(16))) __bf16 v16bf;
typedef __attribute__((ext_vector_type(8)))  float  v8f;

#define S_LEN   2048
#define HID     2048
#define NHEADS  8
#define NKVH    2
#define HDIM    256
#define BATCH   2
#define BS      (BATCH * S_LEN)   // 4096 rows

union Frag16 {
    v16bf v;
    uint4 q[2];
};

// A-operand (16x32 bf16, M x K): lane m (0-15) holds row m, K = {k0..k0+7, k0+16..k0+23};
// lanes 16-31 hold row m with K shifted by +8 (per CDNA5 ISA 7.12.2 layout).
__device__ __forceinline__ v16bf load_a_frag(const bf16_t* base, int ld, int row0, int k0) {
    const int lane = threadIdx.x & 31;
    const int r    = row0 + (lane & 15);
    const int ks   = k0 + ((lane >> 4) << 3);
    const bf16_t* p = base + (size_t)r * ld + ks;
    Frag16 f;
    f.q[0] = *(const uint4*)(p);
    f.q[1] = *(const uint4*)(p + 16);
    return f.v;
}

// B-operand (32x16 bf16, K x N) from row-major W[N,K]: lane n (0-15) holds column n
// (= W row n0+n), K = {k0..k0+7, k0+16..k0+23}; lanes 16-31 hold K shifted by +8.
__device__ __forceinline__ v16bf load_b_frag(const bf16_t* base, int ld, int n0, int k0) {
    const int lane = threadIdx.x & 31;
    const int n    = n0 + (lane & 15);
    const int ks   = k0 + ((lane >> 4) << 3);
    const bf16_t* p = base + (size_t)n * ld + ks;
    Frag16 f;
    f.q[0] = *(const uint4*)(p);
    f.q[1] = *(const uint4*)(p + 16);
    return f.v;
}

__device__ __forceinline__ v8f wmma_bf16(v16bf a, v16bf b, v8f c) {
    return __builtin_amdgcn_wmma_f32_16x16x32_bf16(false, a, false, b, (short)0, c,
                                                   false, false);
}

// ---------------------------------------------------------------------------
// fp32 -> bf16 cast
// ---------------------------------------------------------------------------
__global__ void cast_f32_to_bf16(const float* __restrict__ in, bf16_t* __restrict__ out,
                                 int n) {
    int i = blockIdx.x * 256 + threadIdx.x;
    if (i < n) out[i] = (bf16_t)in[i];
}

// ---------------------------------------------------------------------------
// Generic NT GEMM: C[M,N] (fp32) = A[M,K] (bf16, row-major) x W[N,K]^T (bf16, row-major)
// Block: 256 threads (8 waves). Block tile 128x128; wave tile 32x64 (2x4 WMMA accums):
// 6 fragment fetches feed 8 WMMAs per k-step.
// M % 128 == 0, N % 128 == 0, K % 32 == 0.
// ---------------------------------------------------------------------------
__global__ __launch_bounds__(256) void gemm_nt_bf16(const bf16_t* __restrict__ A,
                                                    const bf16_t* __restrict__ W,
                                                    float* __restrict__ C,
                                                    int M, int N, int K) {
    const int w    = threadIdx.x >> 5;      // 0..7
    const int lane = threadIdx.x & 31;
    const int m0   = blockIdx.y * 128 + (w >> 1) * 32;   // 4 wave-rows of 32
    const int n0   = blockIdx.x * 128 + (w & 1) * 64;    // 2 wave-cols of 64
    (void)M;

    v8f acc[2][4] = {};

    for (int k = 0; k < K; k += 32) {
        v16bf a0 = load_a_frag(A, K, m0,      k);
        v16bf a1 = load_a_frag(A, K, m0 + 16, k);
        v16bf b0 = load_b_frag(W, K, n0,      k);
        v16bf b1 = load_b_frag(W, K, n0 + 16, k);
        v16bf b2 = load_b_frag(W, K, n0 + 32, k);
        v16bf b3 = load_b_frag(W, K, n0 + 48, k);
        acc[0][0] = wmma_bf16(a0, b0, acc[0][0]);
        acc[0][1] = wmma_bf16(a0, b1, acc[0][1]);
        acc[0][2] = wmma_bf16(a0, b2, acc[0][2]);
        acc[0][3] = wmma_bf16(a0, b3, acc[0][3]);
        acc[1][0] = wmma_bf16(a1, b0, acc[1][0]);
        acc[1][1] = wmma_bf16(a1, b1, acc[1][1]);
        acc[1][2] = wmma_bf16(a1, b2, acc[1][2]);
        acc[1][3] = wmma_bf16(a1, b3, acc[1][3]);
    }

    // C/D layout: VGPR r, lanes 0-15 -> row r, col lane; lanes 16-31 -> row r+8.
    const int rowoff = (lane >> 4) << 3;
    const int col    = lane & 15;
    for (int mi = 0; mi < 2; ++mi)
        for (int ni = 0; ni < 4; ++ni)
            for (int r = 0; r < 8; ++r)
                C[(size_t)(m0 + mi * 16 + rowoff + r) * N + (n0 + ni * 16 + col)] =
                    acc[mi][ni][r];
}

// ---------------------------------------------------------------------------
// RoPE: in fp32 [BS, nh*256] -> out bf16 same layout.
// out[d] = x[d]*cos(pos*invfreq[d%128]) + rot[d]*sin(...), rot = concat(-x2, x1).
// ---------------------------------------------------------------------------
__global__ void rope_f32_to_bf16(const float* __restrict__ in, bf16_t* __restrict__ out,
                                 int nh) {
    const int rowlen = nh * HDIM;
    const int total  = BS * rowlen;
    const int i = blockIdx.x * 256 + threadIdx.x;
    if (i >= total) return;
    const int bs  = i / rowlen;
    const int rem = i - bs * rowlen;
    const int d   = rem & (HDIM - 1);
    const int s   = bs & (S_LEN - 1);
    const int fi  = d & 127;
    const float inv_freq = __powf(10000.0f, -(float)(2 * fi) * (1.0f / 256.0f));
    const float ang = (float)s * inv_freq;
    float sn, cs;
    __sincosf(ang, &sn, &cs);
    const float x   = in[i];
    const float rot = (d < 128) ? -in[i + 128] : in[i - 128];
    out[i] = (bf16_t)(x * cs + rot * sn);
}

// ---------------------------------------------------------------------------
// Transpose V: fp32 [B*S, NKVH*HDIM] -> bf16 Vt[B][NKVH][HDIM][S]
// ---------------------------------------------------------------------------
__global__ void v_transpose(const float* __restrict__ V, bf16_t* __restrict__ Vt) {
    const int total = BATCH * NKVH * HDIM * S_LEN;   // 2,097,152
    const int i = blockIdx.x * 256 + threadIdx.x;
    if (i >= total) return;
    const int s = i & (S_LEN - 1);
    int r = i >> 11;
    const int d  = r & (HDIM - 1);
    r >>= 8;
    const int kv = r & (NKVH - 1);
    const int b  = r >> 1;
    Vt[i] = (bf16_t)V[((size_t)(b * S_LEN + s)) * (NKVH * HDIM) + kv * HDIM + d];
}

// ---------------------------------------------------------------------------
// Flash-style attention with tanh softcap.
// Grid: (S/64, NHEADS, BATCH). Block: 128 threads = 4 waves; wave w owns
// 16 query rows [qtile*64 + w*16, +16). Full (non-causal) softmax over S keys.
// ---------------------------------------------------------------------------
__global__ __launch_bounds__(128) void attention_kernel(const bf16_t* __restrict__ Qb,
                                                        const bf16_t* __restrict__ Kb,
                                                        const bf16_t* __restrict__ Vt,
                                                        bf16_t* __restrict__ Ab) {
    __shared__ __align__(32) bf16_t Plds[4][16 * 64];

    const int w    = threadIdx.x >> 5;
    const int lane = threadIdx.x & 31;
    const int qtile = blockIdx.x;
    const int h     = blockIdx.y;
    const int b     = blockIdx.z;
    const int kvh   = h >> 2;                       // GQA groups = 4

    const bf16_t* Qbase = Qb + (size_t)b * S_LEN * (NHEADS * HDIM) + h * HDIM;
    const bf16_t* Kbase = Kb + (size_t)b * S_LEN * (NKVH * HDIM) + kvh * HDIM;
    const bf16_t* Vbase = Vt + ((size_t)(b * NKVH + kvh)) * HDIM * S_LEN;
    const int q0 = qtile * 64 + w * 16;

    // Q fragments for this wave's 16 rows over D=256 (8 k-steps of 32), resident.
    v16bf qa[8];
    for (int ks = 0; ks < 8; ++ks)
        qa[ks] = load_a_frag(Qbase, NHEADS * HDIM, q0, ks * 32);

    v8f oacc[16] = {};                 // 16 x 256 output accumulator (16 d-tiles)
    float mrow[8], lrow[8];
    for (int r = 0; r < 8; ++r) { mrow[r] = -1e30f; lrow[r] = 0.0f; }

    const float kcap = 0.0625f / 50.0f;   // (1/sqrt(256)) / SOFTCAP

    for (int kc = 0; kc < S_LEN; kc += 64) {
        // ---- scores: 16 x 64 tile = 4 accumulators over kj subtiles ----
        v8f sacc[4] = {};
        for (int ks = 0; ks < 8; ++ks) {
            for (int kj = 0; kj < 4; ++kj) {
                v16bf kb = load_b_frag(Kbase, NKVH * HDIM, kc + kj * 16, ks * 32);
                sacc[kj] = wmma_bf16(qa[ks], kb, sacc[kj]);
            }
        }
        // softcap: 50 * tanh(raw / 800)
        for (int kj = 0; kj < 4; ++kj)
            for (int r = 0; r < 8; ++r)
                sacc[kj][r] = 50.0f * tanhf(sacc[kj][r] * kcap);

        // ---- online softmax (rows live in 16-lane halves) ----
        float f[8];
        for (int r = 0; r < 8; ++r) {
            float v = fmaxf(fmaxf(sacc[0][r], sacc[1][r]),
                            fmaxf(sacc[2][r], sacc[3][r]));
            for (int msk = 1; msk < 16; msk <<= 1)
                v = fmaxf(v, __shfl_xor(v, msk, 32));
            const float mnew = fmaxf(mrow[r], v);
            f[r] = __expf(mrow[r] - mnew);
            float ps = 0.0f;
            for (int kj = 0; kj < 4; ++kj) {
                const float p = __expf(sacc[kj][r] - mnew);
                sacc[kj][r] = p;
                ps += p;
            }
            for (int msk = 1; msk < 16; msk <<= 1)
                ps += __shfl_xor(ps, msk, 32);
            lrow[r] = lrow[r] * f[r] + ps;
            mrow[r] = mnew;
        }
        // rescale O
        for (int t = 0; t < 16; ++t)
            for (int r = 0; r < 8; ++r)
                oacc[t][r] *= f[r];

        // ---- P -> LDS (C-layout to memory), re-read in A-layout ----
        const int rowoff = (lane >> 4) << 3;
        const int col    = lane & 15;
        for (int kj = 0; kj < 4; ++kj)
            for (int r = 0; r < 8; ++r)
                Plds[w][(rowoff + r) * 64 + kj * 16 + col] = (bf16_t)sacc[kj][r];

        // ---- O += P(16x64) x V(64x256) ----
        for (int kt = 0; kt < 2; ++kt) {
            v16bf pa = load_a_frag(&Plds[w][0], 64, 0, kt * 32);
            for (int t = 0; t < 16; ++t) {
                v16bf vb = load_b_frag(Vbase, S_LEN, t * 16, kc + kt * 32);
                oacc[t] = wmma_bf16(pa, vb, oacc[t]);
            }
        }
    }

    // ---- normalize and store attn (bf16, [BS, NHEADS*HDIM]) ----
    const int rowoff = (lane >> 4) << 3;
    const int col    = lane & 15;
    float inv[8];
    for (int r = 0; r < 8; ++r) inv[r] = 1.0f / lrow[r];
    for (int t = 0; t < 16; ++t)
        for (int r = 0; r < 8; ++r) {
            const size_t row = (size_t)b * S_LEN + qtile * 64 + w * 16 + rowoff + r;
            Ab[row * (NHEADS * HDIM) + h * HDIM + t * 16 + col] =
                (bf16_t)(oacc[t][r] * inv[r]);
        }
}

// ---------------------------------------------------------------------------
// kernel_launch
// ---------------------------------------------------------------------------
extern "C" void kernel_launch(void* const* d_in, const int* in_sizes, int n_in,
                              void* d_out, int out_size, void* d_ws, size_t ws_size,
                              hipStream_t stream) {
    (void)in_sizes; (void)n_in; (void)out_size; (void)ws_size;

    const float* hidden = (const float*)d_in[0];
    const float* Wq     = (const float*)d_in[1];
    const float* Wk     = (const float*)d_in[2];
    const float* Wv     = (const float*)d_in[3];
    const float* Wo     = (const float*)d_in[4];
    float*       out    = (float*)d_out;

    char* ws = (char*)d_ws;
    size_t off = 0;
    auto alloc = [&](size_t bytes) -> void* {
        void* p = ws + off;
        off = (off + bytes + 255) & ~(size_t)255;
        return p;
    };

    bf16_t* Hb   = (bf16_t*)alloc((size_t)BS * HID * 2);
    bf16_t* Wqb  = (bf16_t*)alloc((size_t)NHEADS * HDIM * HID * 2);
    bf16_t* Wkb  = (bf16_t*)alloc((size_t)NKVH * HDIM * HID * 2);
    bf16_t* Wvb  = (bf16_t*)alloc((size_t)NKVH * HDIM * HID * 2);
    bf16_t* Wob  = (bf16_t*)alloc((size_t)HID * NHEADS * HDIM * 2);
    float*  Qf   = (float*)alloc((size_t)BS * NHEADS * HDIM * 4);
    float*  Kf   = (float*)alloc((size_t)BS * NKVH * HDIM * 4);
    float*  Vf   = (float*)alloc((size_t)BS * NKVH * HDIM * 4);
    bf16_t* Qbf  = (bf16_t*)alloc((size_t)BS * NHEADS * HDIM * 2);
    bf16_t* Kbf  = (bf16_t*)alloc((size_t)BS * NKVH * HDIM * 2);
    bf16_t* Vtb  = (bf16_t*)alloc((size_t)BATCH * NKVH * HDIM * S_LEN * 2);
    bf16_t* Attb = (bf16_t*)alloc((size_t)BS * NHEADS * HDIM * 2);

    auto cast = [&](const float* in, bf16_t* o, int n) {
        cast_f32_to_bf16<<<(n + 255) / 256, 256, 0, stream>>>(in, o, n);
    };
    cast(hidden, Hb,  BS * HID);
    cast(Wq,     Wqb, NHEADS * HDIM * HID);
    cast(Wk,     Wkb, NKVH * HDIM * HID);
    cast(Wv,     Wvb, NKVH * HDIM * HID);
    cast(Wo,     Wob, HID * NHEADS * HDIM);

    const dim3 blk(256);
    // Q/K/V projections: C = H x W^T
    gemm_nt_bf16<<<dim3((NHEADS * HDIM) / 128, BS / 128), blk, 0, stream>>>(
        Hb, Wqb, Qf, BS, NHEADS * HDIM, HID);
    gemm_nt_bf16<<<dim3((NKVH * HDIM) / 128, BS / 128), blk, 0, stream>>>(
        Hb, Wkb, Kf, BS, NKVH * HDIM, HID);
    gemm_nt_bf16<<<dim3((NKVH * HDIM) / 128, BS / 128), blk, 0, stream>>>(
        Hb, Wvb, Vf, BS, NKVH * HDIM, HID);

    // RoPE (Q, K) -> bf16
    rope_f32_to_bf16<<<(BS * NHEADS * HDIM + 255) / 256, 256, 0, stream>>>(Qf, Qbf, NHEADS);
    rope_f32_to_bf16<<<(BS * NKVH * HDIM + 255) / 256, 256, 0, stream>>>(Kf, Kbf, NKVH);

    // V -> [B][KV][D][S] bf16 for contiguous PV B-operand loads
    v_transpose<<<(BATCH * NKVH * HDIM * S_LEN + 255) / 256, 256, 0, stream>>>(Vf, Vtb);

    // attention
    attention_kernel<<<dim3(S_LEN / 64, NHEADS, BATCH), dim3(128), 0, stream>>>(
        Qbf, Kbf, Vtb, Attb);

    // output projection: out = attn x Wo^T (fp32 out)
    gemm_nt_bf16<<<dim3(HID / 128, BS / 128), blk, 0, stream>>>(
        Attb, Wob, out, BS, HID, NHEADS * HDIM);
}
